// multiheadAttentionDirect_66262755442757
// MI455X (gfx1250) — compile-verified, matrix operand order checked by reading, not run on hardware
//
#include <hip/hip_runtime.h>

// Problem constants (fixed by the reference)
#define B_  4
#define N_  2048
#define F_  256
#define H_  8
#define O_  256
#define HF_ 2048   // H*F

typedef __attribute__((ext_vector_type(16))) _Float16 v16h;
typedef __attribute__((ext_vector_type(8)))  _Float16 v8h;
typedef __attribute__((ext_vector_type(8)))  float    v8f;
typedef __attribute__((ext_vector_type(4)))  unsigned int u32x4;
typedef __attribute__((ext_vector_type(8)))  int          i32x8;
typedef __attribute__((ext_vector_type(4)))  int          i32x4;

union Frag { v16h v; v8h h[2]; };

// Load a 16x16x32-f16 WMMA A/B fragment from a row-major operand where the
// lane's matrix row pointer is `rowptr` (already offset to k0).
// ISA 7.12.2: lane<16 holds K={0..7,16..23}, lane>=16 holds K={8..15,24..31}.
__device__ __forceinline__ v16h load_frag_row(const _Float16* rowptr, int lane) {
  const _Float16* p = rowptr + ((lane & 16) ? 8 : 0);
  Frag f;
  f.h[0] = *(const v8h*)(p);
  f.h[1] = *(const v8h*)(p + 16);
  return f.v;
}

#if defined(__has_builtin)
#if __has_builtin(__builtin_amdgcn_tensor_load_to_lds) && \
    __has_builtin(__builtin_amdgcn_s_wait_tensorcnt)
#define HAVE_TDM 1
#endif
#endif

#ifdef HAVE_TDM
// Issue a TDM 2D tile load (global row-major -> LDS packed rows).
// D# packing per CDNA5 ISA 8.3-8.5: group0 = {ctrl, lds_addr, gaddr lo/hi|type},
// group1 = {mask/data_size, tensor dims, tile dims, dim0 stride}.
// This toolchain declares the 6-arg builtin: (g0, g1, g2, g3, g4, cpol).
__device__ __forceinline__ void tdm_load_2d(unsigned int lds_off,
                                            const void* gaddr,
                                            unsigned int tile_d0,   // contiguous elems
                                            unsigned int tile_d1,   // rows
                                            unsigned int stride0) { // elems between rows
  unsigned long long ga = (unsigned long long)(uintptr_t)gaddr;
  u32x4 g0;
  g0[0] = 1u;                                   // count=1 user descriptor
  g0[1] = lds_off;                              // LDS byte address of tile start
  g0[2] = (unsigned int)ga;                     // global_addr[31:0]
  g0[3] = (unsigned int)((ga >> 32) & 0x01FFFFFFu) | (2u << 30); // addr[56:32] | type=2
  i32x8 g1;
  unsigned int td0 = tile_d0, td1 = tile_d1;    // tensor dims == tile (all in-bounds)
  g1[0] = (int)(1u << 16);                      // workgroup_mask=0, data_size=1 (2B)
  g1[1] = (int)((td0 & 0xFFFFu) << 16);         // tensor_dim0[15:0] @ bits 63:48
  g1[2] = (int)((td0 >> 16) | ((td1 & 0xFFFFu) << 16)); // dim0 hi | dim1 lo
  g1[3] = (int)((td1 >> 16) | (tile_d0 << 16)); // dim1 hi | tile_dim0
  g1[4] = (int)(tile_d1 & 0xFFFFu);             // tile_dim1 | tile_dim2=0
  g1[5] = (int)stride0;                         // tensor_dim0_stride[31:0]
  g1[6] = 0;                                    // stride0 hi | dim1_stride lo (unused 2D)
  g1[7] = 0;
  i32x4 z4 = {0, 0, 0, 0};
  i32x8 z8 = {0, 0, 0, 0, 0, 0, 0, 0};
  __builtin_amdgcn_tensor_load_to_lds(g0, g1, z4, z4, z8, 0);
}
#endif

// ---------------------------------------------------------------------------
// Kernel 0: one-time f32 -> f16 packing / transposition into workspace.
// ---------------------------------------------------------------------------
__global__ void k_pack(const float* __restrict__ nodes,
                       const float* __restrict__ Wh,
                       const float* __restrict__ Wo,
                       _Float16* __restrict__ nodes16,
                       _Float16* __restrict__ vt,
                       _Float16* __restrict__ wt,
                       _Float16* __restrict__ wot) {
  int i = blockIdx.x * blockDim.x + threadIdx.x;
  if (i < B_ * N_ * F_) {
    float v = nodes[i];
    nodes16[i] = (_Float16)v;
    int f = i % F_;
    int n = (i / F_) % N_;
    int b = i / (F_ * N_);
    vt[((size_t)b * F_ + f) * N_ + n] = (_Float16)v;
  }
  if (i < H_ * F_ * F_) {
    int g = i % F_;
    int f = (i / F_) % F_;
    int h = i / (F_ * F_);
    wt[((size_t)h * F_ + g) * F_ + f] = (_Float16)Wh[i];
  }
  if (i < HF_ * O_) {
    int o = i % O_;
    int c = i / O_;
    wot[(size_t)o * HF_ + c] = (_Float16)Wo[i];
  }
}

// ---------------------------------------------------------------------------
// Kernel 1: xw[b,h] = nodes[b] @ W_heads[h]  (NT WMMA GEMM)
// ---------------------------------------------------------------------------
__global__ void k_gemm_xw(const _Float16* __restrict__ nodes16,
                          const _Float16* __restrict__ wt,
                          _Float16* __restrict__ xw) {
  int lane = threadIdx.x & 31;
  int wid  = blockIdx.x * (blockDim.x >> 5) + (threadIdx.x >> 5);
  int ntile = wid & 15;
  int mtile = (wid >> 4) & 127;
  int h     = (wid >> 11) & 7;
  int b     = wid >> 14;
  if (b >= B_) return;
  int l15 = lane & 15;
  const _Float16* Arow = nodes16 + ((size_t)b * N_ + mtile * 16 + l15) * F_;
  const _Float16* Brow = wt + ((size_t)h * F_ + ntile * 16 + l15) * F_;
  v8f acc;
#pragma unroll
  for (int i = 0; i < 8; ++i) acc[i] = 0.0f;
#pragma unroll
  for (int k0 = 0; k0 < F_; k0 += 32) {
    v16h a  = load_frag_row(Arow + k0, lane);
    v16h bf = load_frag_row(Brow + k0, lane);
    acc = __builtin_amdgcn_wmma_f32_16x16x32_f16(false, a, false, bf,
                                                 (short)0, acc, false, false);
  }
  int hi8 = (lane & 16) ? 8 : 0;
  _Float16* out = xw + (((size_t)b * H_ + h) * N_ + mtile * 16) * F_ + ntile * 16 + l15;
#pragma unroll
  for (int i = 0; i < 8; ++i)
    out[(size_t)(i + hi8) * F_] = (_Float16)acc[i];
}

// ---------------------------------------------------------------------------
// Kernel 2: fused flash attention. 4 waves / block, 64 query rows / block.
// K/V chunks (32 keys) staged once per block into LDS via TDM (double-
// buffered, overlapped with WMMA compute) instead of per-wave global reads:
// cuts L2 fragment traffic 4x vs one-wave blocks.
// ---------------------------------------------------------------------------
__global__ void __launch_bounds__(128, 1)
k_flash(const _Float16* __restrict__ xw,
        const _Float16* __restrict__ nodes16,
        const _Float16* __restrict__ vt,
        _Float16* __restrict__ msgs) {
  __shared__ _Float16 Klds[2][32 * F_];   // [buf][key][f]   16 KB each
  __shared__ _Float16 Vlds[2][F_ * 32];   // [buf][f][key]   16 KB each
  __shared__ _Float16 Plds[4][16 * 32];   // per-wave P tile bounce buffers

  int tid  = threadIdx.x;
  int lane = tid & 31;
  int w    = tid >> 5;
  int qb64 = blockIdx.x & 31;             // N/64 = 32 query super-blocks
  int h    = (blockIdx.x >> 5) & 7;
  int b    = blockIdx.x >> 8;
  int l15  = lane & 15;
  int hi8  = (lane & 16) ? 8 : 0;
  int qrow0 = qb64 * 64 + w * 16;         // this wave's 16 query rows

  const _Float16* Kbase = nodes16 + (size_t)b * N_ * F_;
  const _Float16* Vbase = vt + (size_t)b * F_ * N_;

  // Resident Q fragments (16 rows x 256 -> 8 fragments, 64 VGPRs)
  const _Float16* Qrow = xw + (((size_t)b * H_ + h) * N_ + qrow0 + l15) * F_;
  v16h qf[8];
#pragma unroll
  for (int kk = 0; kk < 8; ++kk) qf[kk] = load_frag_row(Qrow + kk * 32, lane);

  const float norm = 0.0625f;             // 1/sqrt(256)
  v8f acc[16];
#pragma unroll
  for (int t = 0; t < 16; ++t)
#pragma unroll
    for (int i = 0; i < 8; ++i) acc[t][i] = 0.0f;
  v8f mrun, lrun;
#pragma unroll
  for (int i = 0; i < 8; ++i) { mrun[i] = -1e30f; lrun[i] = 0.0f; }

#ifdef HAVE_TDM
  bool issuer = (w == 0);
  if (issuer) {  // prefetch chunk 0
    tdm_load_2d((unsigned int)(uintptr_t)&Klds[0][0], Kbase, F_, 32, F_);
    tdm_load_2d((unsigned int)(uintptr_t)&Vlds[0][0], Vbase, 32, F_, N_);
  }
#endif

  for (int c = 0; c < N_; c += 32) {
    int cb = (c >> 5) & 1;
#ifdef HAVE_TDM
    if (issuer) __builtin_amdgcn_s_wait_tensorcnt(0);  // chunk c landed in LDS
    __syncthreads();                                   // publish to all waves
    if (issuer && (c + 32) < N_) {                     // prefetch chunk c+1
      int nb = cb ^ 1;
      tdm_load_2d((unsigned int)(uintptr_t)&Klds[nb][0],
                  Kbase + (size_t)(c + 32) * F_, F_, 32, F_);
      tdm_load_2d((unsigned int)(uintptr_t)&Vlds[nb][0],
                  Vbase + (c + 32), 32, F_, N_);
    }
#else
    // Fallback: cooperative synchronous staging by all 128 threads.
    __syncthreads();
    {
      const v8h* gk = (const v8h*)(Kbase + (size_t)c * F_);
      v8h* sk = (v8h*)&Klds[cb][0];
#pragma unroll
      for (int r = 0; r < 8; ++r) sk[tid + 128 * r] = gk[tid + 128 * r];
      v8h* sv = (v8h*)&Vlds[cb][0];
#pragma unroll
      for (int r = 0; r < 8; ++r) {
        int idx = tid + 128 * r;           // idx over 1024 v8h = [f][key8]
        int f = idx >> 2, kq = (idx & 3) * 8;
        sv[idx] = *(const v8h*)(Vbase + (size_t)f * N_ + c + kq);
      }
    }
    __syncthreads();
#endif

    // ---- scores: two 16x16 tiles from LDS-staged K ----
    v8f s0, s1;
#pragma unroll
    for (int i = 0; i < 8; ++i) { s0[i] = 0.0f; s1[i] = 0.0f; }
    const _Float16* K0row = &Klds[cb][(size_t)l15 * F_];
    const _Float16* K1row = &Klds[cb][(size_t)(16 + l15) * F_];
#pragma unroll
    for (int kk = 0; kk < 8; ++kk) {
      v16h b0 = load_frag_row(K0row + kk * 32, lane);
      s0 = __builtin_amdgcn_wmma_f32_16x16x32_f16(false, qf[kk], false, b0,
                                                  (short)0, s0, false, false);
      v16h b1 = load_frag_row(K1row + kk * 32, lane);
      s1 = __builtin_amdgcn_wmma_f32_16x16x32_f16(false, qf[kk], false, b1,
                                                  (short)0, s1, false, false);
    }

    // ---- online softmax update (rows 0..7 in lanes 0..15, 8..15 in 16..31) --
    v8f mc;
#pragma unroll
    for (int i = 0; i < 8; ++i) {
      float a = s0[i] * norm, bb = s1[i] * norm;
      s0[i] = a; s1[i] = bb;
      mc[i] = fmaxf(a, bb);
    }
#pragma unroll
    for (int off = 1; off < 16; off <<= 1)
#pragma unroll
      for (int i = 0; i < 8; ++i)
        mc[i] = fmaxf(mc[i], __shfl_xor(mc[i], off, 32));
    v8f alpha;
#pragma unroll
    for (int i = 0; i < 8; ++i) {
      float mnew = fmaxf(mrun[i], mc[i]);
      alpha[i] = __expf(mrun[i] - mnew);
      mrun[i] = mnew;
      s0[i] = __expf(s0[i] - mnew);
      s1[i] = __expf(s1[i] - mnew);
    }
    v8f rs;
#pragma unroll
    for (int i = 0; i < 8; ++i) rs[i] = s0[i] + s1[i];
#pragma unroll
    for (int off = 1; off < 16; off <<= 1)
#pragma unroll
      for (int i = 0; i < 8; ++i)
        rs[i] += __shfl_xor(rs[i], off, 32);
#pragma unroll
    for (int i = 0; i < 8; ++i) lrun[i] = lrun[i] * alpha[i] + rs[i];
#pragma unroll
    for (int t = 0; t < 16; ++t)
#pragma unroll
      for (int i = 0; i < 8; ++i) acc[t][i] *= alpha[i];

    // ---- bounce P (16x32 f16) through this wave's private LDS region ----
#pragma unroll
    for (int i = 0; i < 8; ++i) {
      Plds[w][(i + hi8) * 32 + l15]      = (_Float16)s0[i];
      Plds[w][(i + hi8) * 32 + 16 + l15] = (_Float16)s1[i];
    }
    asm volatile("s_wait_dscnt 0x0" ::: "memory");  // in-wave store->load ordering
    v16h pf = load_frag_row(&Plds[w][(size_t)l15 * 32], lane);

    // ---- acc += P @ V over the 16 F-tiles (V staged NT in LDS) ----
#pragma unroll
    for (int t = 0; t < 16; ++t) {
      const _Float16* Vrow = &Vlds[cb][(size_t)(t * 16 + l15) * 32];
      v16h vf = load_frag_row(Vrow, lane);
      acc[t] = __builtin_amdgcn_wmma_f32_16x16x32_f16(false, pf, false, vf,
                                                      (short)0, acc[t], false, false);
    }
  }

  // ---- finalize and store in concat layout msgs[b][n][h*F + f] ----
#pragma unroll
  for (int i = 0; i < 8; ++i) lrun[i] = 1.0f / lrun[i];
  _Float16* Mout = msgs + ((size_t)b * N_ + qrow0) * HF_ + h * F_;
#pragma unroll
  for (int t = 0; t < 16; ++t)
#pragma unroll
    for (int i = 0; i < 8; ++i)
      Mout[(size_t)(i + hi8) * HF_ + t * 16 + l15] = (_Float16)(acc[t][i] * lrun[i]);
}

// ---------------------------------------------------------------------------
// Kernel 3: out = msgs @ W_out + bias  (f32 output)
// ---------------------------------------------------------------------------
__global__ void k_gemm_out(const _Float16* __restrict__ msgs,
                           const _Float16* __restrict__ wot,
                           const float* __restrict__ bias,
                           float* __restrict__ out) {
  int lane = threadIdx.x & 31;
  int wid  = blockIdx.x * (blockDim.x >> 5) + (threadIdx.x >> 5);
  int ntile = wid & 15;
  int mtile = wid >> 4;
  if (mtile >= (B_ * N_) / 16) return;
  int l15 = lane & 15;
  const _Float16* Arow = msgs + (size_t)(mtile * 16 + l15) * HF_;
  const _Float16* Brow = wot + (size_t)(ntile * 16 + l15) * HF_;
  v8f acc;
#pragma unroll
  for (int i = 0; i < 8; ++i) acc[i] = 0.0f;
#pragma unroll 8
  for (int k0 = 0; k0 < HF_; k0 += 32) {
    v16h a  = load_frag_row(Arow + k0, lane);
    v16h bf = load_frag_row(Brow + k0, lane);
    acc = __builtin_amdgcn_wmma_f32_16x16x32_f16(false, a, false, bf,
                                                 (short)0, acc, false, false);
  }
  int hi8 = (lane & 16) ? 8 : 0;
  int col = ntile * 16 + l15;
  float bv = bias[col];
  float* Orow = out + (size_t)(mtile * 16) * O_ + col;
#pragma unroll
  for (int i = 0; i < 8; ++i)
    Orow[(size_t)(i + hi8) * O_] = acc[i] + bv;
}

extern "C" void kernel_launch(void* const* d_in, const int* in_sizes, int n_in,
                              void* d_out, int out_size, void* d_ws, size_t ws_size,
                              hipStream_t stream) {
  (void)in_sizes; (void)n_in; (void)out_size; (void)ws_size;
  const float* nodes = (const float*)d_in[0];
  const float* Wh    = (const float*)d_in[1];
  const float* Wo    = (const float*)d_in[2];
  const float* bias  = (const float*)d_in[3];
  float* out = (float*)d_out;

  char* ws = (char*)d_ws;
  size_t off = 0;
  _Float16* nodes16 = (_Float16*)(ws + off); off += (size_t)B_ * N_ * F_ * 2;
  _Float16* vt      = (_Float16*)(ws + off); off += (size_t)B_ * N_ * F_ * 2;
  _Float16* wt      = (_Float16*)(ws + off); off += (size_t)H_ * F_ * F_ * 2;
  _Float16* wot     = (_Float16*)(ws + off); off += (size_t)HF_ * O_ * 2;
  _Float16* xw      = (_Float16*)(ws + off); off += (size_t)B_ * H_ * N_ * F_ * 2;
  _Float16* msgs    = (_Float16*)(ws + off);

  int packN = B_ * N_ * F_;
  k_pack<<<(packN + 255) / 256, 256, 0, stream>>>(nodes, Wh, Wo,
                                                  nodes16, vt, wt, wot);
  k_gemm_xw<<<65536 / 4, 128, 0, stream>>>(nodes16, wt, xw);
  // 4 waves per block, 64 query rows per block, TDM-staged K/V chunks
  k_flash<<<B_ * H_ * (N_ / 64), 128, 0, stream>>>(xw, nodes16, vt, msgs);
  k_gemm_out<<<8192 / 4, 128, 0, stream>>>(msgs, wot, bias, out);
}
